// DeformableTransformerDecoderLayer_Det_2319282340126
// MI455X (gfx1250) — compile-verified, matrix operand order checked by reading, not run on hardware
//
#include <hip/hip_runtime.h>
#include <hip/hip_bf16.h>
#include <math.h>

typedef __attribute__((ext_vector_type(16))) _Float16 v16h;
typedef __attribute__((ext_vector_type(8)))  _Float16 v8h;
typedef __attribute__((ext_vector_type(8)))  float    v8f;
typedef __attribute__((ext_vector_type(4)))  unsigned v4u;
typedef __attribute__((ext_vector_type(8)))  int      v8i;
typedef __attribute__((ext_vector_type(4)))  int      v4i;

#define CAT16(lo,hi) __builtin_shufflevector(lo,hi,0,1,2,3,4,5,6,7,8,9,10,11,12,13,14,15)

#define D_    256
#define NH_   8
#define DH_   32
#define LQ_   1000
#define NB_   8
#define LEN_  21760
#define R_    (NB_*LQ_)      // 8000 query rows
#define VR_   (NB_*LEN_)     // 174080 value rows
#define DFFN_ 1024

// Tensor Data Mover availability (device pass only; host pass + non-TDM
// toolchains fall back to the synchronous LDS staging path).
#if defined(__AMDGCN__) && __has_builtin(__builtin_amdgcn_tensor_load_to_lds)
#define HAVE_TDM 1
#else
#define HAVE_TDM 0
#endif

#if HAVE_TDM
// LDS byte offset of a __shared__ object: generic -> AS(3) ptr -> int.
__device__ __forceinline__ unsigned lds_offset_of(const void* p) {
  return (unsigned)(size_t)(__attribute__((address_space(3))) const void*)p;
}

// Issue one TDM tile load: [tileRows x 32] f16 tile from a row-major
// [rows x K] f16 matrix starting at gsrc, into LDS at ldsoff with
// 8-half (4-DWORD) padding every 32 halfs (16 DWORDs) -> 40-half rows.
__device__ __forceinline__ void tdm_load_tile(const _Float16* gsrc,
                                              unsigned ldsoff,
                                              int rows, int tileRows, int K) {
  unsigned long long ga = (unsigned long long)(size_t)gsrc;
  v4u g0;
  g0[0] = 1u;                                   // count=1 valid descriptor
  g0[1] = ldsoff;                               // lds_addr
  g0[2] = (unsigned)(ga & 0xffffffffu);         // global_addr[31:0]
  g0[3] = (unsigned)((ga >> 32) & 0x1ffffffu)   // global_addr[56:32]
          | (2u << 30);                         // type=2 ("image")
  unsigned d0 = (unsigned)K;                    // tensor_dim0
  unsigned d1 = (unsigned)rows;                 // tensor_dim1
  v8i g1;
  g1[0] = (int)((1u << 16)        // data_size = 1 -> 2 bytes
              | (1u << 20)        // pad_enable
              | (3u << 22)        // pad_interval: 16 DWORDs
              | (3u << 25));      // pad_amount:   4 DWORDs
  g1[1] = (int)((d0 & 0xffffu) << 16);                       // dim0[15:0]
  g1[2] = (int)(((d0 >> 16) & 0xffffu) | ((d1 & 0xffffu) << 16));
  g1[3] = (int)(((d1 >> 16) & 0xffffu) | (32u << 16));       // tile_dim0=32
  g1[4] = (int)(unsigned)tileRows;                           // tile_dim1
  g1[5] = (int)d0;                                           // dim0_stride lo = K
  g1[6] = 0;                                                 // stride hi / dim1_stride
  g1[7] = 0;
  v4i z4 = {0, 0, 0, 0};
#if __clang_major__ >= 23
  v8i z8 = {0, 0, 0, 0, 0, 0, 0, 0};
  __builtin_amdgcn_tensor_load_to_lds(g0, g1, z4, z4, z8, 0);
#else
  __builtin_amdgcn_tensor_load_to_lds(g0, g1, z4, z4, 0);
#endif
}
#endif  // HAVE_TDM

// ---------------------------------------------------------------------------
// f32 -> f16 convert (optionally fused residual add):  o = f16(a [+ b])
// ---------------------------------------------------------------------------
__global__ __launch_bounds__(256) void cvt_add_h(const float* __restrict__ a,
                                                 const float* __restrict__ b,
                                                 _Float16* __restrict__ o,
                                                 size_t n) {
  size_t i = (size_t)blockIdx.x * 256 + threadIdx.x;
  if (i < n) {
    float v = a[i];
    if (b) v += b[i];
    o[i] = (_Float16)v;
  }
}

// ---------------------------------------------------------------------------
// Wave-per-row LayerNorm over 256 cols: out = LN(x + y)*g + b
// ---------------------------------------------------------------------------
__global__ __launch_bounds__(256) void ln_k(const float* __restrict__ x,
                                            const float* __restrict__ y,
                                            const float* __restrict__ g,
                                            const float* __restrict__ b,
                                            float* __restrict__ of,
                                            _Float16* __restrict__ oh,
                                            int rows) {
  int row  = blockIdx.x * 8 + (threadIdx.x >> 5);
  int lane = threadIdx.x & 31;
  if (row >= rows) return;
  const float* xp = x + (size_t)row * D_;
  const float* yp = y + (size_t)row * D_;
  float v[8], s = 0.f, sq = 0.f;
#pragma unroll
  for (int i = 0; i < 8; ++i) {
    int c = lane + 32 * i;
    float t = xp[c] + yp[c];
    v[i] = t; s += t; sq += t * t;
  }
#pragma unroll
  for (int m = 1; m < 32; m <<= 1) {   // wave32 butterfly reduce
    s  += __shfl_xor(s,  m, 32);
    sq += __shfl_xor(sq, m, 32);
  }
  float mean = s * (1.f / D_);
  float var  = sq * (1.f / D_) - mean * mean;
  float rstd = rsqrtf(var + 1e-5f);
#pragma unroll
  for (int i = 0; i < 8; ++i) {
    int c = lane + 32 * i;
    float o = (v[i] - mean) * rstd * g[c] + b[c];
    of[(size_t)row * D_ + c] = o;
    if (oh) oh[(size_t)row * D_ + c] = (_Float16)o;
  }
}

// ---------------------------------------------------------------------------
// WMMA GEMM: C[M x N] = A[M x K](f16) * W[N x K]^T(f16) + bias (f32)
// 4 waves/block; block tile 64(M) x 128(N); K staged 32 at a time through
// LDS, double-buffered via TDM tensor_load_to_lds when available.
// Each wave: 1 A-fragment x 8 B-fragments x 8 v_wmma_f32_16x16x32_f16.
// ---------------------------------------------------------------------------
template <bool OUT_HALF, bool RELU>
__global__ __launch_bounds__(128) void gemm_k(const _Float16* __restrict__ A,
                                              const _Float16* __restrict__ W,
                                              const float* __restrict__ bias,
                                              void* __restrict__ Cv,
                                              int M, int N, int K) {
  __shared__ _Float16 As[2][64][40];    // 40-half rows: 16B-aligned v8h access
  __shared__ _Float16 Bs[2][128][40];
  const int m0 = blockIdx.y * 64, n0 = blockIdx.x * 128;
  const int tid = threadIdx.x;
  const int wv = tid >> 5, lane = tid & 31;
  const int r = lane & 15, half = lane >> 4;

  v8f acc[8] = {};

#if HAVE_TDM
  if (wv == 0) {   // prologue: async tiles for k0=0 into buffer 0
    tdm_load_tile(A + (size_t)m0 * K, lds_offset_of(&As[0][0][0]), M, 64, K);
    tdm_load_tile(W + (size_t)n0 * K, lds_offset_of(&Bs[0][0][0]), N, 128, K);
  }
#endif

  int cur = 0;
  for (int k0 = 0; k0 < K; k0 += 32) {
#if HAVE_TDM
    if (wv == 0) {
      if (k0 + 32 < K) {  // pipeline next K tile into the other buffer
        tdm_load_tile(A + (size_t)m0 * K + (k0 + 32),
                      lds_offset_of(&As[cur ^ 1][0][0]), M, 64, K);
        tdm_load_tile(W + (size_t)n0 * K + (k0 + 32),
                      lds_offset_of(&Bs[cur ^ 1][0][0]), N, 128, K);
        __builtin_amdgcn_s_wait_tensorcnt((short)2);  // current buffer done
      } else {
        __builtin_amdgcn_s_wait_tensorcnt((short)0);
      }
    }
#else
    // synchronous cooperative tile loads (16B per thread-load)
#pragma unroll
    for (int i = 0; i < 2; ++i) {
      int c = tid + 128 * i;
      int row = c >> 2, o = (c & 3) * 8;
      *(v8h*)&As[cur][row][o] = *(const v8h*)(A + (size_t)(m0 + row) * K + k0 + o);
    }
#pragma unroll
    for (int i = 0; i < 4; ++i) {
      int c = tid + 128 * i;
      int row = c >> 2, o = (c & 3) * 8;
      *(v8h*)&Bs[cur][row][o] = *(const v8h*)(W + (size_t)(n0 + row) * K + k0 + o);
    }
    if (k0 + 32 < K) {
      __builtin_prefetch(A + (size_t)(m0 + (tid >> 1)) * K + k0 + 32, 0, 1);
      __builtin_prefetch(W + (size_t)(n0 + (tid >> 1)) * K + k0 + 32, 0, 1);
    }
#endif
    __syncthreads();

    // A fragment: lane holds row r; K = [8*half, +8) U [16+8*half, +8)
    v8h alo = *(const v8h*)&As[cur][16 * wv + r][8 * half];
    v8h ahi = *(const v8h*)&As[cur][16 * wv + r][16 + 8 * half];
    v16h a = CAT16(alo, ahi);
#pragma unroll
    for (int nt = 0; nt < 8; ++nt) {
      v8h blo = *(const v8h*)&Bs[cur][nt * 16 + r][8 * half];
      v8h bhi = *(const v8h*)&Bs[cur][nt * 16 + r][16 + 8 * half];
      v16h bf = CAT16(blo, bhi);
      acc[nt] = __builtin_amdgcn_wmma_f32_16x16x32_f16(
          false, a, false, bf, (short)0, acc[nt], false, false);
    }
    __syncthreads();
    cur ^= 1;
  }

  // epilogue: C/D layout row = g + 8*half, col = lane&15
#pragma unroll
  for (int nt = 0; nt < 8; ++nt) {
#pragma unroll
    for (int g = 0; g < 8; ++g) {
      int m   = m0 + 16 * wv + g + 8 * half;
      int col = n0 + nt * 16 + r;
      float vo = acc[nt][g] + (bias ? bias[col] : 0.f);
      if (RELU) vo = fmaxf(vo, 0.f);
      if (OUT_HALF) ((_Float16*)Cv)[(size_t)m * N + col] = (_Float16)vo;
      else          ((float*)Cv)[(size_t)m * N + col] = vo;
    }
  }
}

// ---------------------------------------------------------------------------
// Flash self-attention: one wave handles a 16-query block of one (n, head).
// qk holds [n,lq,512] f16 (q cols 0..255, k cols 256..511, head h at h*32).
// vh holds [n,lq,256] f16. Output f16 [n,lq,256].
// ---------------------------------------------------------------------------
__global__ __launch_bounds__(32) void attn_flash(const _Float16* __restrict__ qk,
                                                 const _Float16* __restrict__ vh,
                                                 _Float16* __restrict__ outh) {
  __shared__ float    Sb[16][32];
  __shared__ _Float16 Pb[16][32];
  __shared__ _Float16 Vt[32][40];        // [dh][key], 80B row stride
  __shared__ float mSt[16], lSt[16], aSt[16];

  const int qb = blockIdx.x, h = blockIdx.y, n = blockIdx.z;
  const int lane = threadIdx.x;
  const int r = lane & 15, half = lane >> 4;
  const float scale = 0.17677669529663687f;  // 1/sqrt(32)

  v16h qa = {};
  {
    int m = qb * 16 + r;
    if (m < LQ_) {
      const _Float16* qp = qk + ((size_t)(n * LQ_ + m)) * 512 + h * 32;
      v8h lo = *(const v8h*)(qp + 8 * half);
      v8h hi = *(const v8h*)(qp + 16 + 8 * half);
      qa = CAT16(lo, hi);
    }
  }
  if (lane < 16) { mSt[lane] = -1e30f; lSt[lane] = 0.f; }
  __syncthreads();

  v8f o0 = {}, o1 = {};

  for (int kb = 0; kb < 32; ++kb) {      // 32 x 32 keys = 1024 (>=1000)
    // stage V tile transposed into LDS: Vt[dh][key_local]
    {
      int key = kb * 32 + lane;
      if (key < LQ_) {
        const _Float16* vp = vh + ((size_t)(n * LQ_ + key)) * D_ + h * 32;
#pragma unroll
        for (int d = 0; d < 32; ++d) Vt[d][lane] = vp[d];
      } else {
#pragma unroll
        for (int d = 0; d < 32; ++d) Vt[d][lane] = (_Float16)0.f;
      }
    }
    // S = Q * K^T for two 16-key sub-tiles (DH=32 == WMMA K)
#pragma unroll
    for (int sub = 0; sub < 2; ++sub) {
      int kc = kb * 32 + sub * 16 + r;
      v16h kf = {};
      if (kc < LQ_) {
        const _Float16* kp = qk + ((size_t)(n * LQ_ + kc)) * 512 + 256 + h * 32;
        v8h lo = *(const v8h*)(kp + 8 * half);
        v8h hi = *(const v8h*)(kp + 16 + 8 * half);
        kf = CAT16(lo, hi);
      }
      v8f cz = {};
      v8f s = __builtin_amdgcn_wmma_f32_16x16x32_f16(
          false, qa, false, kf, (short)0, cz, false, false);
#pragma unroll
      for (int g = 0; g < 8; ++g) {
        float sc = s[g] * scale;
        if (kc >= LQ_) sc = -1e30f;
        Sb[g + 8 * half][sub * 16 + r] = sc;
      }
    }
    __syncthreads();

    // online softmax (16 lanes, one per query row)
    if (lane < 16) {
      float tmax = -1e30f;
#pragma unroll
      for (int j = 0; j < 32; ++j) tmax = fmaxf(tmax, Sb[lane][j]);
      float mo = mSt[lane];
      float nm = fmaxf(mo, tmax);
      float al = __expf(mo - nm);
      float ls = 0.f;
#pragma unroll
      for (int j = 0; j < 32; ++j) {
        float p = __expf(Sb[lane][j] - nm);
        Pb[lane][j] = (_Float16)p;
        ls += p;
      }
      mSt[lane] = nm;
      lSt[lane] = lSt[lane] * al + ls;
      aSt[lane] = al;
    }
    __syncthreads();

    // rescale accumulators, then O += P * V  (two 16-wide N tiles of DH=32)
#pragma unroll
    for (int g = 0; g < 8; ++g) {
      float a_ = aSt[g + 8 * half];
      o0[g] *= a_; o1[g] *= a_;
    }
    v8h plo = *(const v8h*)&Pb[r][8 * half];
    v8h phi = *(const v8h*)&Pb[r][16 + 8 * half];
    v16h pa = CAT16(plo, phi);
    v8h v0lo = *(const v8h*)&Vt[r][8 * half];
    v8h v0hi = *(const v8h*)&Vt[r][16 + 8 * half];
    v16h vb0 = CAT16(v0lo, v0hi);
    v8h v1lo = *(const v8h*)&Vt[16 + r][8 * half];
    v8h v1hi = *(const v8h*)&Vt[16 + r][16 + 8 * half];
    v16h vb1 = CAT16(v1lo, v1hi);
    o0 = __builtin_amdgcn_wmma_f32_16x16x32_f16(false, pa, false, vb0, (short)0, o0, false, false);
    o1 = __builtin_amdgcn_wmma_f32_16x16x32_f16(false, pa, false, vb1, (short)0, o1, false, false);
    __syncthreads();
  }

#pragma unroll
  for (int g = 0; g < 8; ++g) {
    int mloc = g + 8 * half;
    int m = qb * 16 + mloc;
    if (m < LQ_) {
      float inv = 1.f / lSt[mloc];
      _Float16* op = outh + ((size_t)(n * LQ_ + m)) * D_ + h * 32;
      op[r]      = (_Float16)(o0[g] * inv);
      op[16 + r] = (_Float16)(o1[g] * inv);
    }
  }
}

// ---------------------------------------------------------------------------
// Multi-scale deformable sampling. One wave per (n, lq, h); lane = channel.
// Fuses the 16-way attention-weight softmax.
// ---------------------------------------------------------------------------
__global__ __launch_bounds__(256) void ms_deform_k(const _Float16* __restrict__ val,
                                                   const float* __restrict__ offs,
                                                   const float* __restrict__ awr,
                                                   const float* __restrict__ refp,
                                                   _Float16* __restrict__ out) {
  int task = blockIdx.x * 8 + (threadIdx.x >> 5);
  int lane = threadIdx.x & 31;
  int n   = task / (LQ_ * NH_);
  int rem = task % (LQ_ * NH_);
  int lq  = rem / NH_;
  int h   = rem % NH_;
  size_t row = (size_t)n * LQ_ + lq;

  const float* aw = awr  + row * 128 + h * 16;
  const float* of = offs + row * 256 + h * 32;
  const float* rp = refp + row * 8;

  float w[16];
  float mx = -1e30f;
#pragma unroll
  for (int j = 0; j < 16; ++j) { w[j] = aw[j]; mx = fmaxf(mx, w[j]); }
  float sm = 0.f;
#pragma unroll
  for (int j = 0; j < 16; ++j) { w[j] = __expf(w[j] - mx); sm += w[j]; }
  float inv = 1.f / sm;

  const int Hs[4] = {128, 64, 32, 16};
  const int Ws[4] = {128, 64, 32, 16};
  const int St[4] = {0, 16384, 20480, 21504};

  float acc = 0.f;
#pragma unroll
  for (int l = 0; l < 4; ++l) {
    int Wl = Ws[l], Hl = Hs[l], st = St[l];
    float rx = rp[l * 2], ry = rp[l * 2 + 1];
#pragma unroll
    for (int p = 0; p < 4; ++p) {
      int j = l * 4 + p;
      float locx = rx + of[j * 2 + 0] / (float)Wl;
      float locy = ry + of[j * 2 + 1] / (float)Hl;
      float x = locx * Wl - 0.5f;
      float y = locy * Hl - 0.5f;
      float x0f = floorf(x), y0f = floorf(y);
      int x0 = (int)x0f, y0 = (int)y0f;
      float fx = x - x0f, fy = y - y0f;
      float s = 0.f;
#pragma unroll
      for (int dy = 0; dy < 2; ++dy) {
#pragma unroll
        for (int dx = 0; dx < 2; ++dx) {
          float wt = (dy ? fy : 1.f - fy) * (dx ? fx : 1.f - fx);
          int xi = x0 + dx, yi = y0 + dy;
          if (xi >= 0 && xi < Wl && yi >= 0 && yi < Hl) {
            size_t idx = ((size_t)n * LEN_ + st + (size_t)yi * Wl + xi) * D_ + h * 32 + lane;
            s += wt * (float)val[idx];
          }
        }
      }
      acc += w[j] * inv * s;
    }
  }
  out[row * D_ + h * 32 + lane] = (_Float16)acc;
}

// ---------------------------------------------------------------------------
// Orchestration
// ---------------------------------------------------------------------------
extern "C" void kernel_launch(void* const* d_in, const int* in_sizes, int n_in,
                              void* d_out, int out_size, void* d_ws, size_t ws_size,
                              hipStream_t stream) {
  (void)in_sizes; (void)n_in; (void)out_size; (void)ws_size;
  const float* tgt      = (const float*)d_in[0];
  const float* qpos     = (const float*)d_in[1];
  const float* refp     = (const float*)d_in[2];
  const float* src      = (const float*)d_in[3];
  const float* sa_in_w  = (const float*)d_in[6];
  const float* sa_in_b  = (const float*)d_in[7];
  const float* sa_out_w = (const float*)d_in[8];
  const float* sa_out_b = (const float*)d_in[9];
  const float* n2g = (const float*)d_in[10]; const float* n2b = (const float*)d_in[11];
  const float* off_w = (const float*)d_in[12]; const float* off_b = (const float*)d_in[13];
  const float* aw_w  = (const float*)d_in[14]; const float* aw_b  = (const float*)d_in[15];
  const float* val_w = (const float*)d_in[16]; const float* val_b = (const float*)d_in[17];
  const float* outp_w = (const float*)d_in[18]; const float* outp_b = (const float*)d_in[19];
  const float* n1g = (const float*)d_in[20]; const float* n1b = (const float*)d_in[21];
  const float* f1w = (const float*)d_in[22]; const float* f1b = (const float*)d_in[23];
  const float* f2w = (const float*)d_in[24]; const float* f2b = (const float*)d_in[25];
  const float* n3g = (const float*)d_in[26]; const float* n3b = (const float*)d_in[27];

  char* base = (char*)d_ws;
  size_t cur = 0;
  auto alloc = [&](size_t bytes) -> void* {
    void* p = base + cur;
    cur = (cur + bytes + 255) & ~(size_t)255;
    return p;
  };

  // f16 weights
  _Float16* hw_sain = (_Float16*)alloc((size_t)768 * 256 * 2);
  _Float16* hw_saout = (_Float16*)alloc((size_t)256 * 256 * 2);
  _Float16* hw_off  = (_Float16*)alloc((size_t)256 * 256 * 2);
  _Float16* hw_aw   = (_Float16*)alloc((size_t)128 * 256 * 2);
  _Float16* hw_val  = (_Float16*)alloc((size_t)256 * 256 * 2);
  _Float16* hw_outp = (_Float16*)alloc((size_t)256 * 256 * 2);
  _Float16* hw_f1   = (_Float16*)alloc((size_t)1024 * 256 * 2);
  _Float16* hw_f2   = (_Float16*)alloc((size_t)256 * 1024 * 2);
  // activations
  _Float16* h_qk   = (_Float16*)alloc((size_t)R_ * 256 * 2);
  _Float16* h_tgt  = (_Float16*)alloc((size_t)R_ * 256 * 2);
  _Float16* qk_out = (_Float16*)alloc((size_t)R_ * 512 * 2);
  _Float16* v_h    = (_Float16*)alloc((size_t)R_ * 256 * 2);
  _Float16* att_h  = (_Float16*)alloc((size_t)R_ * 256 * 2);
  float*    sa_f   = (float*)alloc((size_t)R_ * 256 * 4);
  float*    tgt2   = (float*)alloc((size_t)R_ * 256 * 4);
  _Float16* h_query = (_Float16*)alloc((size_t)R_ * 256 * 2);
  _Float16* src_h  = (_Float16*)alloc((size_t)VR_ * 256 * 2);
  _Float16* value_h = (_Float16*)alloc((size_t)VR_ * 256 * 2);
  float*    off_f  = (float*)alloc((size_t)R_ * 256 * 4);
  float*    aw_f   = (float*)alloc((size_t)R_ * 128 * 4);
  _Float16* ca_h   = (_Float16*)alloc((size_t)R_ * 256 * 2);
  float*    ca_f   = (float*)alloc((size_t)R_ * 256 * 4);
  float*    tgt3   = (float*)alloc((size_t)R_ * 256 * 4);
  _Float16* h_tgt3 = (_Float16*)alloc((size_t)R_ * 256 * 2);
  _Float16* ffn1_h = (_Float16*)alloc((size_t)R_ * 1024 * 2);
  float*    ffn2_f = (float*)alloc((size_t)R_ * 256 * 4);

  auto cvt = [&](const float* a, const float* b, _Float16* o, size_t n) {
    unsigned blocks = (unsigned)((n + 255) / 256);
    cvt_add_h<<<blocks, 256, 0, stream>>>(a, b, o, n);
  };

  // weight conversions
  cvt(sa_in_w,  nullptr, hw_sain, (size_t)768 * 256);
  cvt(sa_out_w, nullptr, hw_saout, (size_t)256 * 256);
  cvt(off_w,    nullptr, hw_off, (size_t)256 * 256);
  cvt(aw_w,     nullptr, hw_aw, (size_t)128 * 256);
  cvt(val_w,    nullptr, hw_val, (size_t)256 * 256);
  cvt(outp_w,   nullptr, hw_outp, (size_t)256 * 256);
  cvt(f1w,      nullptr, hw_f1, (size_t)1024 * 256);
  cvt(f2w,      nullptr, hw_f2, (size_t)256 * 1024);
  // activations
  cvt(tgt, qpos, h_qk, (size_t)R_ * 256);
  cvt(tgt, nullptr, h_tgt, (size_t)R_ * 256);
  cvt(src, nullptr, src_h, (size_t)VR_ * 256);

  // q,k projection (fused, N=512)
  gemm_k<true, false><<<dim3(512 / 128, R_ / 64), 128, 0, stream>>>(
      h_qk, hw_sain, sa_in_b, (void*)qk_out, R_, 512, 256);
  // v projection
  gemm_k<true, false><<<dim3(256 / 128, R_ / 64), 128, 0, stream>>>(
      h_tgt, hw_sain + (size_t)512 * 256, sa_in_b + 512, (void*)v_h, R_, 256, 256);
  // flash self-attention
  attn_flash<<<dim3(63, NH_, NB_), 32, 0, stream>>>(qk_out, v_h, att_h);
  // SA output projection
  gemm_k<false, false><<<dim3(256 / 128, R_ / 64), 128, 0, stream>>>(
      att_h, hw_saout, sa_out_b, (void*)sa_f, R_, 256, 256);
  // LN2
  ln_k<<<R_ / 8, 256, 0, stream>>>(tgt, sa_f, n2g, n2b, tgt2, (_Float16*)nullptr, R_);
  // query = tgt2 + qpos (f16)
  cvt(tgt2, qpos, h_query, (size_t)R_ * 256);
  // value projection over all spatial tokens
  gemm_k<true, false><<<dim3(256 / 128, VR_ / 64), 128, 0, stream>>>(
      src_h, hw_val, val_b, (void*)value_h, VR_, 256, 256);
  // sampling offsets + attention weights
  gemm_k<false, false><<<dim3(256 / 128, R_ / 64), 128, 0, stream>>>(
      h_query, hw_off, off_b, (void*)off_f, R_, 256, 256);
  gemm_k<false, false><<<dim3(128 / 128, R_ / 64), 128, 0, stream>>>(
      h_query, hw_aw, aw_b, (void*)aw_f, R_, 128, 256);
  // deformable sampling
  ms_deform_k<<<(NB_ * LQ_ * NH_) / 8, 256, 0, stream>>>(value_h, off_f, aw_f, refp, ca_h);
  // cross-attn output projection
  gemm_k<false, false><<<dim3(256 / 128, R_ / 64), 128, 0, stream>>>(
      ca_h, hw_outp, outp_b, (void*)ca_f, R_, 256, 256);
  // LN1 (also emits f16 for FFN input)
  ln_k<<<R_ / 8, 256, 0, stream>>>(tgt2, ca_f, n1g, n1b, tgt3, h_tgt3, R_);
  // FFN
  gemm_k<true, true><<<dim3(DFFN_ / 128, R_ / 64), 128, 0, stream>>>(
      h_tgt3, hw_f1, f1b, (void*)ffn1_h, R_, DFFN_, 256);
  gemm_k<false, false><<<dim3(256 / 128, R_ / 64), 128, 0, stream>>>(
      ffn1_h, hw_f2, f2b, (void*)ffn2_f, R_, 256, DFFN_);
  // LN3 -> final output
  ln_k<<<R_ / 8, 256, 0, stream>>>(tgt3, ffn2_f, n3g, n3b, (float*)d_out,
                                   (_Float16*)nullptr, R_);
}
